// AAConv_56994216018444
// MI455X (gfx1250) — matrix operand order repaired and checked by reading, not compiled, not run on hardware
//
#include <hip/hip_runtime.h>
#include <hip/hip_bf16.h>

typedef __attribute__((ext_vector_type(16))) __bf16 v16bf;
typedef __attribute__((ext_vector_type(8)))  float  v8f;

#define CIN   256
#define HW    1024
#define DHEAD 32
#define QSCALE 0.17677669529663687f   // 32^-0.5

union Pack16 { uint4 u4[2]; v16bf v; };

// A 16x32 bf16 layout: lane needs K chunks [8*hi, 8*hi+7] and [16+8*hi, 23+8*hi]
// B 32x16 bf16 layout: lane needs K chunks [8*hi, 8*hi+7] and [16+8*hi, 23+8*hi]
// Both contiguous when LDS tiles are stored [row=M/N][col=K] (rows padded to 40).

__device__ __forceinline__ v16bf load_A(const __bf16* row, int hi) {
  Pack16 p;
  p.u4[0] = *(const uint4*)(row + 8 * hi);
  p.u4[1] = *(const uint4*)(row + 16 + 8 * hi);
  return p.v;
}
__device__ __forceinline__ v16bf load_B(const __bf16* row, int hi) {
  Pack16 p;
  p.u4[0] = *(const uint4*)(row + 8 * hi);
  p.u4[1] = *(const uint4*)(row + 16 + 8 * hi);
  return p.v;
}
// pack 8 floats -> 8 bf16 -> one ds_store_b128
__device__ __forceinline__ void store8_bf16(__bf16* dst, const float* f) {
  union { __bf16 h[8]; uint4 u; } tmp;
#pragma unroll
  for (int i = 0; i < 8; ++i) tmp.h[i] = (__bf16)f[i];
  *(uint4*)dst = tmp.u;
}

// gfx1250 async DMA: each lane copies 16B global -> LDS, tracked by ASYNCcnt.
__device__ __forceinline__ void async_b128(void* lds_dst, const void* gsrc) {
  uint32_t l = (uint32_t)(uintptr_t)lds_dst;
  uint64_t g = (uint64_t)(uintptr_t)gsrc;
  asm volatile("global_load_async_to_lds_b128 %0, %1, off"
               :: "v"(l), "v"(g) : "memory");
}
__device__ __forceinline__ void wait_async0() {
  asm volatile("s_wait_asynccnt 0x0" ::: "memory");
}

#define WMMA_BF16(A, B, C) \
  __builtin_amdgcn_wmma_f32_16x16x32_bf16(false, (A), false, (B), (short)0, (C), false, false)

// ---------------------------------------------------------------------------
// Kernel 1: kqv projection (bf16 WMMA GEMM).
//   q -> wsQ transposed  [b*8+h][n][32]   (pre-scaled by 32^-0.5)
//   k -> wsK transposed  [b*8+h][n][32]
//   v -> wsV chan-major  [b*256+c][n]
// WG 256 thr, tile 64j x 128n; wave tile 32x32 (4 WMMA / 8 ds_load_b128).
// ---------------------------------------------------------------------------
__global__ __launch_bounds__(256) void kqv_kernel(
    const float* __restrict__ x, const float* __restrict__ w,
    const float* __restrict__ bias, __bf16* __restrict__ wsQ,
    __bf16* __restrict__ wsK, __bf16* __restrict__ wsV) {
  __shared__ __bf16 sA[64][40];    // [j][k]
  __shared__ __bf16 sB[128][40];   // [n][k]
  const int n0 = blockIdx.x * 128;
  const int j0 = blockIdx.y * 64;
  const int b  = blockIdx.z;
  const int t = threadIdx.x;
  const int wave = t >> 5, lane = t & 31;
  const int wj = wave >> 2, wn = wave & 3;          // 2 x 4 waves
  const int lm = lane & 15, hi = lane >> 4;
  const float* xb = x + (size_t)b * CIN * HW;

  v8f acc[2][2] = {};
  for (int c0 = 0; c0 < CIN; c0 += 32) {
    __syncthreads();
    {   // A tile: w[c, j] -> sA[j][k], one b128 store per thread
      int jj = t & 63, kk = (t >> 6) << 3;
      float f[8];
#pragma unroll
      for (int i = 0; i < 8; ++i) f[i] = w[(size_t)(c0 + kk + i) * 768 + j0 + jj];
      store8_bf16(&sA[jj][kk], f);
    }
    {   // B tile: x[c, n] -> sB[n][k], two b128 stores per thread
      int nn = t & 127, kk = (t >> 7) << 4;
      float f[16];
#pragma unroll
      for (int i = 0; i < 16; ++i) f[i] = xb[(size_t)(c0 + kk + i) * HW + n0 + nn];
      store8_bf16(&sB[nn][kk], f);
      store8_bf16(&sB[nn][kk + 8], f + 8);
    }
    __syncthreads();
    v16bf a0 = load_A(&sA[wj * 32 + lm][0], hi);
    v16bf a1 = load_A(&sA[wj * 32 + 16 + lm][0], hi);
    v16bf b0 = load_B(&sB[wn * 32 + lm][0], hi);
    v16bf b1 = load_B(&sB[wn * 32 + 16 + lm][0], hi);
    acc[0][0] = WMMA_BF16(a0, b0, acc[0][0]);
    acc[0][1] = WMMA_BF16(a0, b1, acc[0][1]);
    acc[1][0] = WMMA_BF16(a1, b0, acc[1][0]);
    acc[1][1] = WMMA_BF16(a1, b1, acc[1][1]);
  }
  if (j0 < 512) {   // k or q region -> transposed [bh][n][32]
    const bool isQ = (j0 >= 256);
    const int hh = ((isQ ? j0 - 256 : j0) >> 5) + wj;   // head (j tile < 32 within)
    __bf16* base = (isQ ? wsQ : wsK) + (size_t)(b * 8 + hh) * HW * 32;
#pragma unroll
    for (int js = 0; js < 2; ++js)
#pragma unroll
      for (int ns = 0; ns < 2; ++ns) {
        int n = n0 + wn * 32 + ns * 16 + lm;
        __bf16* dst = base + (size_t)n * 32;
#pragma unroll
        for (int r = 0; r < 8; ++r) {
          int d = js * 16 + r + 8 * hi;
          float v = acc[js][ns][r] + bias[j0 + wj * 32 + d];
          if (isQ) v *= QSCALE;
          dst[d] = (__bf16)v;
        }
      }
  } else {          // v region -> channel-major [b*256+c][n]
#pragma unroll
    for (int js = 0; js < 2; ++js)
#pragma unroll
      for (int ns = 0; ns < 2; ++ns) {
        int n = n0 + wn * 32 + ns * 16 + lm;
#pragma unroll
        for (int r = 0; r < 8; ++r) {
          int jloc = wj * 32 + js * 16 + r + 8 * hi;
          int c = (j0 - 512) + jloc;
          wsV[((size_t)b * 256 + c) * HW + n] =
              (__bf16)(acc[js][ns][r] + bias[j0 + jloc]);
        }
      }
  }
}

// ---------------------------------------------------------------------------
// Kernel 2: 3x3 SAME conv as implicit GEMM over 9 taps, K=2304 in chunks of 32.
// WG tile 64co x 128n; wave tile 32x32.
// ---------------------------------------------------------------------------
__global__ __launch_bounds__(256) void conv_kernel(
    const float* __restrict__ x, const float* __restrict__ w,
    const float* __restrict__ bias, float* __restrict__ out) {
  __shared__ __bf16 sA[64][40];    // [co][k]
  __shared__ __bf16 sB[128][40];   // [n][k]
  const int n0  = blockIdx.x * 128;
  const int co0 = blockIdx.y * 64;
  const int b   = blockIdx.z;
  const int t = threadIdx.x;
  const int wave = t >> 5, lane = t & 31;
  const int wj = wave >> 2, wn = wave & 3;
  const int lm = lane & 15, hi = lane >> 4;
  const float* xb = x + (size_t)b * CIN * HW;

  v8f acc[2][2] = {};
  for (int tap = 0; tap < 9; ++tap) {
    const int dy = tap / 3 - 1, dx = tap % 3 - 1;
    for (int c0 = 0; c0 < CIN; c0 += 32) {
      __syncthreads();
      {   // A tile: conv_w[tap, c, co] -> sA[co][k]
        int jj = t & 63, kk = (t >> 6) << 3;
        float f[8];
#pragma unroll
        for (int i = 0; i < 8; ++i)
          f[i] = w[((size_t)tap * 256 + (c0 + kk + i)) * 256 + co0 + jj];
        store8_bf16(&sA[jj][kk], f);
      }
      {   // B tile: shifted x with zero fill -> sB[n][k] (validity uniform in k)
        int nn = t & 127, kk = (t >> 7) << 4;
        int n = n0 + nn;
        int hh = (n >> 5) + dy, ww = (n & 31) + dx;
        bool ok = ((unsigned)hh < 32u) && ((unsigned)ww < 32u);
        const float* src = xb + (hh << 5) + ww;
        float f[16];
#pragma unroll
        for (int i = 0; i < 16; ++i)
          f[i] = ok ? src[(size_t)(c0 + kk + i) * HW] : 0.f;
        store8_bf16(&sB[nn][kk], f);
        store8_bf16(&sB[nn][kk + 8], f + 8);
      }
      __syncthreads();
      v16bf a0 = load_A(&sA[wj * 32 + lm][0], hi);
      v16bf a1 = load_A(&sA[wj * 32 + 16 + lm][0], hi);
      v16bf b0 = load_B(&sB[wn * 32 + lm][0], hi);
      v16bf b1 = load_B(&sB[wn * 32 + 16 + lm][0], hi);
      acc[0][0] = WMMA_BF16(a0, b0, acc[0][0]);
      acc[0][1] = WMMA_BF16(a0, b1, acc[0][1]);
      acc[1][0] = WMMA_BF16(a1, b0, acc[1][0]);
      acc[1][1] = WMMA_BF16(a1, b1, acc[1][1]);
    }
  }
#pragma unroll
  for (int js = 0; js < 2; ++js)
#pragma unroll
    for (int ns = 0; ns < 2; ++ns) {
      int n = n0 + wn * 32 + ns * 16 + lm;
#pragma unroll
      for (int r = 0; r < 8; ++r) {
        int co = co0 + wj * 32 + js * 16 + r + 8 * hi;
        out[((size_t)b * 512 + co) * HW + n] = acc[js][ns][r] + bias[co];
      }
    }
}

// ---------------------------------------------------------------------------
// Kernel 3: fused flash attention per (b, head, 128-query block).
// All staging is pure bf16 byte copy -> async DMA to LDS (ASYNCcnt path).
// ---------------------------------------------------------------------------
__global__ __launch_bounds__(256) void attn_kernel(
    const __bf16* __restrict__ wsQ, const __bf16* __restrict__ wsK,
    const __bf16* __restrict__ wsV, float* __restrict__ out) {
  __shared__ __bf16 sQ[128][40];       // [qrow][d]
  __shared__ __bf16 sK[256][40];       // [key][d]
  __shared__ __bf16 sV[32][264];       // [dv][key]
  __shared__ __bf16 sP[8][16][40];     // per-wave P staging [qrow][key]
  const int n0 = blockIdx.x * 128;
  const int h  = blockIdx.y;
  const int b  = blockIdx.z;
  const int t = threadIdx.x;
  const int wave = t >> 5, lane = t & 31;
  const int lm = lane & 15, hi = lane >> 4;

  const __bf16* qT = wsQ + (size_t)(b * 8 + h) * HW * 32;      // [n][32]
  const __bf16* kT = wsK + (size_t)(b * 8 + h) * HW * 32;      // [n][32]
  const __bf16* vC = wsV + ((size_t)b * 256 + h * DHEAD) * HW; // [dv][n]

  {   // stage Q: 128 rows x 64B, 2 async b128 chunks per thread
    int row = t >> 1, half = (t & 1) << 4;
    const __bf16* src = qT + (size_t)(n0 + row) * 32 + half;
    async_b128(&sQ[row][half + 0], src + 0);
    async_b128(&sQ[row][half + 8], src + 8);
  }
  wait_async0();
  __syncthreads();

  v16bf aq = load_A(&sQ[wave * 16 + lm][0], hi);

  float mrow[8], lrow[8];
  v8f o0 = {}, o1 = {};
#pragma unroll
  for (int r = 0; r < 8; ++r) { mrow[r] = -1e30f; lrow[r] = 0.f; }

  for (int kt = 0; kt < 4; ++kt) {
    __syncthreads();
    {   // stage K tile: 256 key rows x 64B, one row per thread
      const __bf16* src = kT + (size_t)(kt * 256 + t) * 32;
#pragma unroll
      for (int i = 0; i < 4; ++i)
        async_b128(&sK[t][8 * i], src + 8 * i);
    }
    {   // stage V tile: 32 dv rows x 256 keys
      int d = t >> 3, seg = (t & 7) << 5;
      const __bf16* src = vC + (size_t)d * HW + kt * 256 + seg;
#pragma unroll
      for (int i = 0; i < 4; ++i)
        async_b128(&sV[d][seg + 8 * i], src + 8 * i);
      if (kt + 1 < 4)
        __builtin_prefetch(kT + (size_t)((kt + 1) * 256 + t) * 32, 0, 1);
    }
    wait_async0();
    __syncthreads();

    for (int kc = 0; kc < 256; kc += 32) {
      // --- S = q^T K for 32 keys: two 16-key WMMAs (K-dim = 32 = d) ---
      v16bf b0 = load_B(&sK[kc + lm][0], hi);
      v16bf b1 = load_B(&sK[kc + 16 + lm][0], hi);
      v8f z = {};
      v8f s0 = WMMA_BF16(aq, b0, z);
      v8f s1 = WMMA_BF16(aq, b1, z);
      // --- online softmax on C-layout rows (16-lane group reductions) ---
#pragma unroll
      for (int r = 0; r < 8; ++r) {
        float mx = fmaxf(s0[r], s1[r]);
#pragma unroll
        for (int off = 8; off >= 1; off >>= 1) mx = fmaxf(mx, __shfl_xor(mx, off, 32));
        float mnew = fmaxf(mrow[r], mx);
        float rescale = __expf(mrow[r] - mnew);
        mrow[r] = mnew;
        float p0 = __expf(s0[r] - mnew);
        float p1 = __expf(s1[r] - mnew);
        float rs = p0 + p1;
#pragma unroll
        for (int off = 8; off >= 1; off >>= 1) rs += __shfl_xor(rs, off, 32);
        lrow[r] = lrow[r] * rescale + rs;
        o0[r] *= rescale;
        o1[r] *= rescale;
        sP[wave][r + 8 * hi][lm]      = (__bf16)p0;   // C-layout -> [qrow][key]
        sP[wave][r + 8 * hi][16 + lm] = (__bf16)p1;
      }
      // --- P back in A-layout (two contiguous b128 chunks per lane) ---
      v16bf ap = load_A(&sP[wave][lm][0], hi);
      // --- O += P @ V^T : B[k=key][n=dv], dv split into two 16-col tiles ---
      v16bf vb0 = load_B(&sV[lm][kc], hi);
      v16bf vb1 = load_B(&sV[16 + lm][kc], hi);
      o0 = WMMA_BF16(ap, vb0, o0);
      o1 = WMMA_BF16(ap, vb1, o1);
    }
  }
  // final normalize + scatter to out channels 256 + h*32 + dv
#pragma unroll
  for (int r = 0; r < 8; ++r) {
    float inv = 1.f / lrow[r];
    int n = n0 + wave * 16 + r + 8 * hi;
    out[((size_t)b * 512 + 256 + h * DHEAD + lm) * HW + n]      = o0[r] * inv;
    out[((size_t)b * 512 + 256 + h * DHEAD + 16 + lm) * HW + n] = o1[r] * inv;
  }
}

extern "C" void kernel_launch(void* const* d_in, const int* in_sizes, int n_in,
                              void* d_out, int out_size, void* d_ws, size_t ws_size,
                              hipStream_t stream) {
  const float* x      = (const float*)d_in[0];
  const float* conv_w = (const float*)d_in[1];
  const float* conv_b = (const float*)d_in[2];
  const float* attn_w = (const float*)d_in[3];
  const float* attn_b = (const float*)d_in[4];
  float* out = (float*)d_out;

  const size_t headElems = (size_t)8 * 8 * 1024 * 32;   // 2,097,152
  __bf16* wsQ = (__bf16*)d_ws;            // q transposed  (4 MB)
  __bf16* wsK = wsQ + headElems;          // k transposed  (4 MB)
  __bf16* wsV = wsK + headElems;          // v chan-major  (4 MB)

  kqv_kernel<<<dim3(8, 12, 8), 256, 0, stream>>>(x, attn_w, attn_b, wsQ, wsK, wsV);
  conv_kernel<<<dim3(8, 4, 8), 256, 0, stream>>>(x, conv_w, conv_b, out);
  attn_kernel<<<dim3(8, 8, 8), 256, 0, stream>>>(wsQ, wsK, wsV, out);
}
